// AtomGCN_71184787964269
// MI455X (gfx1250) — compile-verified, matrix operand order checked by reading, not run on hardware
//
#include <hip/hip_runtime.h>
#include <hip/hip_bf16.h>

// ---------------------------------------------------------------------------
// Problem constants (match reference)
// ---------------------------------------------------------------------------
#define NQ   8192      // query nodes
#define NE   262144    // edges
#define NM   4096      // SA centers (NQ/2)
#define KSA  32        // SA neighbors
#define SA_R2 144.0f   // R_SA^2
#define EPS_BN 1e-5f

typedef _Float16 half8   __attribute__((ext_vector_type(8)));
typedef _Float16 half16  __attribute__((ext_vector_type(16)));
typedef float    floatx8 __attribute__((ext_vector_type(8)));

// ---------------------------------------------------------------------------
// WMMA helpers (CDNA5 wave32, 16x16x32 f16 -> f32)
// Fragment layout per ISA: lane L (row/col = L&15), kb = (L>>4)*8:
//   elems 0..7  = K kb..kb+7,  elems 8..15 = K kb+16..kb+23
// => fragment = two 16-byte loads from a [*, K]-stride row (LDS or global).
// ---------------------------------------------------------------------------
__device__ __forceinline__ half16 frag_pair(half8 lo, half8 hi) {
  return __builtin_shufflevector(lo, hi, 0,1,2,3,4,5,6,7,8,9,10,11,12,13,14,15);
}

__device__ __forceinline__ floatx8 wmma_f16(half16 a, half16 b, floatx8 c) {
  return __builtin_amdgcn_wmma_f32_16x16x32_f16(false, a, false, b, (short)0, c,
                                                false, false);
}

// ---------------------------------------------------------------------------
// A-gather functors
// ---------------------------------------------------------------------------
struct DenseA {
  const float* A; int K;
  __device__ float operator()(int row, int f) const {
    return A[(size_t)row * K + f];
  }
};

struct MggA {   // edge features: [src_x[src](32) | x[tgt](64) | lrf[tgt]@rel(3)]
  const float* x; const float* srcx; const float* pos; const float* spos;
  const float* lrf; const int* src; const int* tgt;
  __device__ float operator()(int e, int f) const {
    if (f < 32) return srcx[(size_t)src[e] * 32 + f];
    if (f < 96) return x[(size_t)tgt[e] * 64 + (f - 32)];
    if (f < 99) {
      int s = src[e], t = tgt[e];
      float d0 = spos[s*3+0] - pos[t*3+0];
      float d1 = spos[s*3+1] - pos[t*3+1];
      float d2 = spos[s*3+2] - pos[t*3+2];
      const float* L = lrf + (size_t)t * 9 + (f - 96) * 3;
      return L[0]*d0 + L[1]*d1 + L[2]*d2;
    }
    return 0.f;
  }
};

struct SaA {    // row = m*32+k: [x0[nbr](128) | lrf[2m]@(pos[nbr]-pos[2m])(3)]
  const float* x0; const float* pos; const float* lrf; const int* nbr;
  __device__ float operator()(int row, int f) const {
    int j = nbr[row];
    if (f < 128) return x0[(size_t)j * 128 + f];
    if (f < 131) {
      int c = (row >> 5) * 2;   // center = 2*m
      float d0 = pos[j*3+0] - pos[c*3+0];
      float d1 = pos[j*3+1] - pos[c*3+1];
      float d2 = pos[j*3+2] - pos[c*3+2];
      const float* L = lrf + (size_t)c * 9 + (f - 128) * 3;
      return L[0]*d0 + L[1]*d1 + L[2]*d2;
    }
    return 0.f;
  }
};

// ---------------------------------------------------------------------------
// Weight pre-conversion: W[ktrue][ncol] f32 -> Wt[ncol][kpad] f16 (zero-padded)
// Done once per call; makes B-fragments two 16B global loads (WGP$-resident).
// ---------------------------------------------------------------------------
__global__ void wconv(const float* __restrict__ W, _Float16* __restrict__ o,
                      int ktrue, int kpad, int ncol) {
  int gid = blockIdx.x * blockDim.x + threadIdx.x;
  if (gid >= ncol * kpad) return;
  int n = gid / kpad, k = gid % kpad;
  o[gid] = (k < ktrue) ? (_Float16)W[(size_t)k * ncol + n] : (_Float16)0.f;
}

// ---------------------------------------------------------------------------
// Generic WMMA GEMM: out[row][n] = sum_k A(row,k) * Wt16[n][k] + bias[n]
// Block = 128 threads (4 waves); each wave owns a private 16-row A strip in
// LDS (no cross-wave sharing -> no block barriers; per-wave DS ordering is
// sufficient). KPAD is compile-time so every B-fragment load is a single
// global_load_b128 with a constant immediate offset off one per-lane base
// pointer (no 64-bit address math, no spills). All 8 B-fragments of a group
// are loaded (clause) before the back-to-back WMMA chain.
// Grid.x = rows/64.
// ---------------------------------------------------------------------------
template <int NT, int KCH, class AG>
__global__ void wmma_gemm(AG ag, const _Float16* __restrict__ Wt,
                          const float* __restrict__ bias, float* __restrict__ out) {
  static_assert(NT % 8 == 0, "NT must be a multiple of 8");
  constexpr int NCOL = NT * 16;
  constexpr int KPAD = KCH * 32;
  __shared__ __align__(16) _Float16 As[4][16 * 32];   // per-wave A strip [row][k]

  const int tid  = threadIdx.x;
  const int lane = tid & 31;
  const int wv   = tid >> 5;
  const int row0 = blockIdx.x * 64 + wv * 16;
  const int ncl  = lane & 15;
  const int kb   = (lane >> 4) * 8;

  floatx8 acc[NT];
#pragma unroll
  for (int t = 0; t < NT; t++)
#pragma unroll
    for (int r = 0; r < 8; r++) acc[t][r] = 0.f;

  // lane converts 16 consecutive K elems of row (lane&15): wide LDS stores,
  // and (for dense A) vectorizable global loads.
  const int ar  = lane & 15;
  const int ak0 = (lane >> 4) * 16;

  // single per-lane B base pointer; all tile/chunk deltas are constants
  const _Float16* wl = Wt + (size_t)ncl * KPAD + kb;

  for (int kc = 0; kc < KCH; kc++) {
    const int k0 = kc * 32;
    half8 sv0, sv1;
#pragma unroll
    for (int i = 0; i < 8; i++) sv0[i] = (_Float16)ag(row0 + ar, k0 + ak0 + i);
#pragma unroll
    for (int i = 0; i < 8; i++) sv1[i] = (_Float16)ag(row0 + ar, k0 + ak0 + 8 + i);
    *(half8*)&As[wv][ar * 32 + ak0]     = sv0;
    *(half8*)&As[wv][ar * 32 + ak0 + 8] = sv1;

    const _Float16* arow = &As[wv][ncl * 32];
    half16 af = frag_pair(*(const half8*)(arow + kb),
                          *(const half8*)(arow + kb + 16));

#pragma unroll
    for (int g = 0; g < NT / 8; g++) {
      half16 bf[8];
#pragma unroll
      for (int u = 0; u < 8; u++) {
        const _Float16* p = wl + (size_t)((g * 8 + u) * 16) * KPAD;  // const offs
        bf[u] = frag_pair(*(const half8*)p, *(const half8*)(p + 16));
      }
#pragma unroll
      for (int u = 0; u < 8; u++)
        acc[g * 8 + u] = wmma_f16(af, bf[u], acc[g * 8 + u]);
    }
    wl += 32;   // next K chunk
  }

  // epilogue: C/D layout -> VGPR r holds row r (lanes 0-15) / row 8+r (16-31)
  const int rbase = row0 + ((lane >> 4) * 8);
#pragma unroll
  for (int t = 0; t < NT; t++) {
    int n = t * 16 + ncl;
    float bv = bias[n];
#pragma unroll
    for (int r = 0; r < 8; r++)
      out[(size_t)(rbase + r) * NCOL + n] = acc[t][r] + bv;
  }
}

// ---------------------------------------------------------------------------
// Column statistics: stats[c] += sum, stats[cols+c] += sumsq  (cols = pow2)
// ---------------------------------------------------------------------------
__global__ void kstats(const float* __restrict__ src, int rows, int cols,
                       float* __restrict__ stats) {
  int g = blockIdx.x * blockDim.x + threadIdx.x;
  int total = gridDim.x * blockDim.x;
  int c = g & (cols - 1);
  int R = total / cols;
  float s = 0.f, q = 0.f;
  for (int r = g / cols; r < rows; r += R) {
    float v = src[(size_t)r * cols + c];
    s += v; q += v * v;
  }
  atomicAdd(&stats[c], s);
  atomicAdd(&stats[cols + c], q);
}

// In-place BatchNorm(train, biased var) + ReLU
__global__ void bnrelu(float* __restrict__ buf, int rows, int cols,
                       const float* __restrict__ stats,
                       const float* __restrict__ g, const float* __restrict__ b) {
  int total = rows * cols;
  for (int i = blockIdx.x * blockDim.x + threadIdx.x; i < total;
       i += gridDim.x * blockDim.x) {
    int c = i & (cols - 1);
    float mean = stats[c] / rows;
    float var  = stats[cols + c] / rows - mean * mean;
    float v = (buf[i] - mean) * rsqrtf(var + EPS_BN) * g[c] + b[c];
    buf[i] = v > 0.f ? v : 0.f;
  }
}

__global__ void kzero(float* __restrict__ p, size_t n) {
  size_t i = (size_t)blockIdx.x * blockDim.x + threadIdx.x;
  size_t st = (size_t)gridDim.x * blockDim.x;
  for (; i < n; i += st) p[i] = 0.f;
}

// ---------------------------------------------------------------------------
// MGG: fused BN+ReLU + segment-max scatter (atomicMax on uint bits; h>=0)
// One thread per (edge, 4 cols). agg must be zero-initialized.
// ---------------------------------------------------------------------------
__global__ void mgg_scatter(const float* __restrict__ h, const int* __restrict__ tgt,
                            const float* __restrict__ stats,
                            const float* __restrict__ g, const float* __restrict__ b,
                            unsigned int* __restrict__ agg) {
  int gid = blockIdx.x * blockDim.x + threadIdx.x;   // NE*32 threads
  int e = gid >> 5, c0 = (gid & 31) * 4;
  int t = tgt[e];
  const float rowsf = (float)NE;
#pragma unroll
  for (int j = 0; j < 4; j++) {
    int c = c0 + j;
    float mean = stats[c] / rowsf;
    float var  = stats[128 + c] / rowsf - mean * mean;
    float v = (h[(size_t)e * 128 + c] - mean) * rsqrtf(var + EPS_BN) * g[c] + b[c];
    v = v > 0.f ? v : 0.f;
    atomicMax(&agg[(size_t)t * 128 + c], __float_as_uint(v));
  }
}

// ---------------------------------------------------------------------------
// SA KNN: one wave per center; d2 cached in LDS; 32 min+argmin selection passes
// ---------------------------------------------------------------------------
__global__ void sa_knn(const float* __restrict__ pos, int* __restrict__ nbr,
                       float* __restrict__ mask) {
  __shared__ __align__(16) float ds[NQ];
  const int m = blockIdx.x;
  const int lane = threadIdx.x;
  const int c = 2 * m;
  const float cx = pos[c*3+0], cy = pos[c*3+1], cz = pos[c*3+2];
  for (int p = lane; p < NQ; p += 32) {
    float dx = pos[p*3+0] - cx, dy = pos[p*3+1] - cy, dz = pos[p*3+2] - cz;
    ds[p] = dx*dx + dy*dy + dz*dz;
  }
  __syncthreads();
  for (int k = 0; k < KSA; k++) {
    float best = 3.0e38f; int bi = 0x7fffffff;
    for (int p = lane; p < NQ; p += 32) {
      float d = ds[p];
      if (d < best || (d == best && p < bi)) { best = d; bi = p; }
    }
    // wave min+argmin reduction (tie -> lower index, mimics stable top_k)
#pragma unroll
    for (int off = 16; off > 0; off >>= 1) {
      float od = __shfl_xor(best, off, 32);
      int   oi = __shfl_xor(bi,   off, 32);
      if (od < best || (od == best && oi < bi)) { best = od; bi = oi; }
    }
    if (lane == 0) {
      nbr[m * KSA + k]  = bi;
      mask[m * KSA + k] = (best <= SA_R2) ? 1.f : 0.f;
      ds[bi] = 3.0e38f;   // exclude from later passes
    }
    __syncthreads();
  }
}

// SA: fused BN+ReLU + mask + max over KSA neighbors. One thread per (m, col).
__global__ void sa_maxbn(const float* __restrict__ h, const float* __restrict__ mask,
                         const float* __restrict__ stats,
                         const float* __restrict__ g, const float* __restrict__ b,
                         float* __restrict__ agg) {
  int gid = blockIdx.x * blockDim.x + threadIdx.x;   // NM*128 threads
  int m = gid >> 7, c = gid & 127;
  const float rowsf = (float)(NM * KSA);
  float mean = stats[c] / rowsf;
  float var  = stats[128 + c] / rowsf - mean * mean;
  float sc = rsqrtf(var + EPS_BN) * g[c];
  float bb = b[c];
  float best = 0.f;
  for (int k = 0; k < KSA; k++) {
    float v = (h[((size_t)(m * KSA + k)) * 128 + c] - mean) * sc + bb;
    v = v > 0.f ? v : 0.f;
    v *= mask[m * KSA + k];
    best = fmaxf(best, v);
  }
  agg[gid] = best;
}

// ---------------------------------------------------------------------------
// FP: 3-NN of each query among the NM centers (register insertion sort)
// ---------------------------------------------------------------------------
__global__ void fp_knn(const float* __restrict__ pos, int* __restrict__ idx3,
                       float* __restrict__ w3) {
  int n = blockIdx.x * blockDim.x + threadIdx.x;
  if (n >= NQ) return;
  float px = pos[n*3+0], py = pos[n*3+1], pz = pos[n*3+2];
  float d0 = 3.0e38f, d1 = 3.0e38f, d2 = 3.0e38f;
  int i0 = 0, i1 = 0, i2 = 0;
  for (int m = 0; m < NM; m++) {
    int c = 2 * m;
    float dx = pos[c*3+0] - px, dy = pos[c*3+1] - py, dz = pos[c*3+2] - pz;
    float d = dx*dx + dy*dy + dz*dz;
    if (d < d0)      { d2 = d1; i2 = i1; d1 = d0; i1 = i0; d0 = d; i0 = m; }
    else if (d < d1) { d2 = d1; i2 = i1; d1 = d;  i1 = m; }
    else if (d < d2) { d2 = d;  i2 = m; }
  }
  idx3[n*3+0] = i0; idx3[n*3+1] = i1; idx3[n*3+2] = i2;
  w3[n*3+0] = 1.f / fmaxf(d0, 1e-16f);
  w3[n*3+1] = 1.f / fmaxf(d1, 1e-16f);
  w3[n*3+2] = 1.f / fmaxf(d2, 1e-16f);
}

// yc[n] = [ knn_interpolate(x1)(256) | x0(128) ]  -> [NQ, 384]
__global__ void fp_concat(const float* __restrict__ x1, const float* __restrict__ x0,
                          const int* __restrict__ idx3, const float* __restrict__ w3,
                          float* __restrict__ yc) {
  int gid = blockIdx.x * blockDim.x + threadIdx.x;   // NQ*384
  int n = gid / 384, c = gid % 384;
  if (c < 256) {
    float w0 = w3[n*3+0], w1 = w3[n*3+1], w2 = w3[n*3+2];
    float s = w0 * x1[(size_t)idx3[n*3+0] * 256 + c]
            + w1 * x1[(size_t)idx3[n*3+1] * 256 + c]
            + w2 * x1[(size_t)idx3[n*3+2] * 256 + c];
    yc[gid] = s / (w0 + w1 + w2);
  } else {
    yc[gid] = x0[(size_t)n * 128 + (c - 256)];
  }
}

// ---------------------------------------------------------------------------
// Host launcher
// ---------------------------------------------------------------------------
extern "C" void kernel_launch(void* const* d_in, const int* in_sizes, int n_in,
                              void* d_out, int out_size, void* d_ws, size_t ws_size,
                              hipStream_t stream) {
  (void)in_sizes; (void)n_in; (void)out_size; (void)ws_size;

  const float* x       = (const float*)d_in[0];
  const float* pos     = (const float*)d_in[1];
  const float* lrf     = (const float*)d_in[2];
  const float* src_x   = (const float*)d_in[3];
  const float* src_pos = (const float*)d_in[4];
  const int*   eidx    = (const int*)  d_in[5];
  const float* mgg_lw  = (const float*)d_in[6];
  const float* mgg_lb  = (const float*)d_in[7];
  const float* mgg_lg  = (const float*)d_in[8];
  const float* mgg_lbt = (const float*)d_in[9];
  const float* mgg_gw  = (const float*)d_in[10];
  const float* mgg_gb  = (const float*)d_in[11];
  const float* mgg_gg  = (const float*)d_in[12];
  const float* mgg_gbt = (const float*)d_in[13];
  const float* sa_lw   = (const float*)d_in[14];
  const float* sa_lb   = (const float*)d_in[15];
  const float* sa_lg   = (const float*)d_in[16];
  const float* sa_lbt  = (const float*)d_in[17];
  const float* sa_gw   = (const float*)d_in[18];
  const float* sa_gb   = (const float*)d_in[19];
  const float* sa_gg   = (const float*)d_in[20];
  const float* sa_gbt  = (const float*)d_in[21];
  const float* fp_w    = (const float*)d_in[22];
  const float* fp_b    = (const float*)d_in[23];
  const float* fp_g    = (const float*)d_in[24];
  const float* fp_bt   = (const float*)d_in[25];

  const int* e_src = eidx;
  const int* e_tgt = eidx + NE;
  float* out = (float*)d_out;

  // workspace bump allocator (256B aligned)
  char* ws = (char*)d_ws;
  size_t off = 0;
  auto walloc = [&](size_t bytes) -> char* {
    char* p = ws + off;
    off += (bytes + 255) & ~(size_t)255;
    return p;
  };
  float* h_mgg   = (float*)walloc((size_t)NE * 128 * 4);       // 134 MB
  float* agg_mgg = (float*)walloc((size_t)NQ * 128 * 4);       // 4 MB (uint bits)
  float* x0      = (float*)walloc((size_t)NQ * 128 * 4);       // 4 MB
  float* h_sa    = (float*)walloc((size_t)NM * KSA * 128 * 4); // 67 MB
  float* agg_sa  = (float*)walloc((size_t)NM * 128 * 4);       // 2 MB
  float* x1      = (float*)walloc((size_t)NM * 256 * 4);       // 4 MB
  float* yc      = (float*)walloc((size_t)NQ * 384 * 4);       // 12.6 MB
  int*   nbr_sa  = (int*)  walloc((size_t)NM * KSA * 4);
  float* mask_sa = (float*)walloc((size_t)NM * KSA * 4);
  int*   idx3    = (int*)  walloc((size_t)NQ * 3 * 4);
  float* w3      = (float*)walloc((size_t)NQ * 3 * 4);
  float* stats   = (float*)walloc((size_t)5 * 512 * 4);
  // pre-converted f16 weights, transposed [ncol][kpad]
  _Float16* w16_mggl = (_Float16*)walloc((size_t)128 * 128 * 2);  // k 99->128
  _Float16* w16_mggg = (_Float16*)walloc((size_t)128 * 128 * 2);  // k 128
  _Float16* w16_sal  = (_Float16*)walloc((size_t)128 * 160 * 2);  // k 131->160
  _Float16* w16_sag  = (_Float16*)walloc((size_t)256 * 128 * 2);  // k 128
  _Float16* w16_fp   = (_Float16*)walloc((size_t)128 * 384 * 2);  // k 384
  float* s0 = stats, *s1 = stats + 512, *s2 = stats + 1024;
  float* s3 = stats + 1536, *s4 = stats + 2048;

  // re-init accumulators every call (graph-replay safe)
  kzero<<<16, 256, 0, stream>>>(stats, 5 * 512);
  kzero<<<2048, 256, 0, stream>>>(agg_mgg, (size_t)NQ * 128);

  // one-shot weight conversion (transpose + pad + f32->f16)
  wconv<<<(128*128 + 255)/256, 256, 0, stream>>>(mgg_lw, w16_mggl,  99, 128, 128);
  wconv<<<(128*128 + 255)/256, 256, 0, stream>>>(mgg_gw, w16_mggg, 128, 128, 128);
  wconv<<<(128*160 + 255)/256, 256, 0, stream>>>(sa_lw,  w16_sal,  131, 160, 128);
  wconv<<<(256*128 + 255)/256, 256, 0, stream>>>(sa_gw,  w16_sag,  128, 128, 256);
  wconv<<<(128*384 + 255)/256, 256, 0, stream>>>(fp_w,   w16_fp,   384, 384, 128);

  // ---- MGG block ----
  wmma_gemm<8, 4, MggA><<<NE / 64, 128, 0, stream>>>(
      MggA{x, src_x, pos, src_pos, lrf, e_src, e_tgt}, w16_mggl, mgg_lb, h_mgg);
  kstats<<<128, 256, 0, stream>>>(h_mgg, NE, 128, s0);
  mgg_scatter<<<NE * 32 / 256, 256, 0, stream>>>(h_mgg, e_tgt, s0, mgg_lg, mgg_lbt,
                                                 (unsigned int*)agg_mgg);
  wmma_gemm<8, 4, DenseA><<<NQ / 64, 128, 0, stream>>>(
      DenseA{agg_mgg, 128}, w16_mggg, mgg_gb, x0);
  kstats<<<128, 256, 0, stream>>>(x0, NQ, 128, s1);
  bnrelu<<<1024, 256, 0, stream>>>(x0, NQ, 128, s1, mgg_gg, mgg_gbt);

  // ---- SA block ----
  sa_knn<<<NM, 32, 0, stream>>>(pos, nbr_sa, mask_sa);
  wmma_gemm<8, 5, SaA><<<NM * KSA / 64, 128, 0, stream>>>(
      SaA{x0, pos, lrf, nbr_sa}, w16_sal, sa_lb, h_sa);
  kstats<<<128, 256, 0, stream>>>(h_sa, NM * KSA, 128, s2);
  sa_maxbn<<<NM * 128 / 256, 256, 0, stream>>>(h_sa, mask_sa, s2, sa_lg, sa_lbt,
                                               agg_sa);
  wmma_gemm<16, 4, DenseA><<<NM / 64, 128, 0, stream>>>(
      DenseA{agg_sa, 128}, w16_sag, sa_gb, x1);
  kstats<<<128, 256, 0, stream>>>(x1, NM, 256, s3);
  bnrelu<<<1024, 256, 0, stream>>>(x1, NM, 256, s3, sa_gg, sa_gbt);

  // ---- FP block ----
  fp_knn<<<NQ / 256, 256, 0, stream>>>(pos, idx3, w3);
  fp_concat<<<NQ * 384 / 256, 256, 0, stream>>>(x1, x0, idx3, w3, yc);
  wmma_gemm<8, 12, DenseA><<<NQ / 64, 128, 0, stream>>>(
      DenseA{yc, 384}, w16_fp, fp_b, out);
  kstats<<<128, 256, 0, stream>>>(out, NQ, 128, s4);
  bnrelu<<<1024, 256, 0, stream>>>(out, NQ, 128, s4, fp_g, fp_bt);
}
